// AttentionLSTMDecoder_24008867184974
// MI455X (gfx1250) — matrix operand order repaired and checked by reading, not compiled, not run on hardware
//
#include <hip/hip_runtime.h>
#include <hip/hip_bf16.h>

// ---------------------------------------------------------------------------
// AttentionLSTMDecoder for MI455X (gfx1250, wave32, WMMA + TDM)
//
// Roofline: dominant cost is streaming the K/V caches every decode step:
//   bf16 K+V = 2 * 128000*512*2B = 262 MB/step * 100 steps = 26 GB
//   => ~1.1 ms at 23.3 TB/s HBM; 262 MB > 192 MB L2 so it streams.
// All GEMMs (~0.9 TFLOP total) are negligible on the bf16 WMMA path, so K/V
// are written bf16 directly from the GEMM epilogue.  The attention score
// pass (the K stream) uses the Tensor Data Mover to double-buffer K tiles
// into LDS (with TDM padding for conflict-free reads), synchronized with
// s_wait_tensorcnt.
// ---------------------------------------------------------------------------

typedef __attribute__((ext_vector_type(16))) __bf16 v16bf;
typedef __attribute__((ext_vector_type(8)))  float  v8f;
typedef unsigned int tdm_u32x4 __attribute__((ext_vector_type(4)));
typedef int          tdm_i32x8 __attribute__((ext_vector_type(8)));
typedef int          tdm_i32x4 __attribute__((ext_vector_type(4)));

#define TM 64
#define TN 64
#define TK 32
#define LDSK 36   // padded bf16 K-stride: 72B rows (8B aligned, conflict-free)

#define GEMM_STORE_BF16 1
#define GEMM_ACCUM      2
#define GEMM_RELU       4

// C[M,N] = A[M,K] @ W[N,K]^T (+bias +bias2); optional accumulate/relu/bf16.
// A,W f32 in global, converted to bf16 while staging to LDS via b128 loads
// (all K used are multiples of 4; rows are 16B aligned).
// Block = 128 threads = 4 waves; each wave owns a 16x64 strip (4 WMMA accs).
__global__ __launch_bounds__(128)
void gemm_wmma(const float* __restrict__ A,
               const float* __restrict__ W,
               const float* __restrict__ bias,
               const float* __restrict__ bias2,
               float* __restrict__ Cf,
               __bf16* __restrict__ Cb,
               int M, int N, int K, int flags)
{
    __shared__ __bf16 As[TM * LDSK];
    __shared__ __bf16 Bs[TN * LDSK];

    const int tid  = threadIdx.x;
    const int lane = tid & 31;
    const int wid  = tid >> 5;      // wave 0..3
    const int half = lane >> 4;     // 0/1
    const int l16  = lane & 15;

    const int m0 = blockIdx.y * TM;
    const int n0 = blockIdx.x * TN;

    v8f acc[4];
    if (flags & GEMM_ACCUM) {
#pragma unroll
        for (int t = 0; t < 4; ++t)
#pragma unroll
            for (int r = 0; r < 8; ++r) {
                int row = m0 + wid * 16 + r + 8 * half;
                int col = n0 + t * 16 + l16;
                acc[t][r] = Cf[(size_t)row * N + col];
            }
    } else {
        v8f z = {0.f,0.f,0.f,0.f,0.f,0.f,0.f,0.f};
#pragma unroll
        for (int t = 0; t < 4; ++t) acc[t] = z;
    }

    const int KT = (K + TK - 1) / TK;
    for (int kb = 0; kb < KT; ++kb) {
        const int k0 = kb * TK;
        // stage 64x32 A-tile and 64x32 W-tile: b128 loads, f32 -> bf16
#pragma unroll
        for (int i = 0; i < 4; ++i) {
            int idx4 = tid + i * 128;        // 0..511 float4 slots
            int r    = idx4 >> 3;            // row 0..63
            int c4   = (idx4 & 7) << 2;      // col 0,4,..,28
            int gk   = k0 + c4;
            float4 a = make_float4(0.f, 0.f, 0.f, 0.f);
            float4 b = make_float4(0.f, 0.f, 0.f, 0.f);
            if (gk < K) {                    // K % 4 == 0 -> no straddle
                a = *(const float4*)(A + (size_t)(m0 + r) * K + gk);
                b = *(const float4*)(W + (size_t)(n0 + r) * K + gk);
            }
            __bf16* as = &As[r * LDSK + c4];
            __bf16* bs = &Bs[r * LDSK + c4];
            as[0] = (__bf16)a.x; as[1] = (__bf16)a.y;
            as[2] = (__bf16)a.z; as[3] = (__bf16)a.w;
            bs[0] = (__bf16)b.x; bs[1] = (__bf16)b.y;
            bs[2] = (__bf16)b.z; bs[3] = (__bf16)b.w;
        }
        __syncthreads();

        // A fragment: 16x32 bf16, ISA 7.12.2 layout
        v16bf af;
        {
            const __bf16* ar = &As[(wid * 16 + l16) * LDSK];
#pragma unroll
            for (int j = 0; j < 8; ++j) {
                int k = ((j < 4) ? (2 * j) : (16 + 2 * (j - 4))) + 8 * half;
                af[2 * j]     = ar[k];
                af[2 * j + 1] = ar[k + 1];
            }
        }
#pragma unroll
        for (int t = 0; t < 4; ++t) {
            // B fragment: 32x16, same k order as A (lanes 0-15: K 0..7,16..23; lanes 16-31: K 8..15,24..31)
            v16bf bf;
            const __bf16* br = &Bs[(t * 16 + l16) * LDSK];
#pragma unroll
            for (int j = 0; j < 8; ++j) {
                int k = ((j < 4) ? (2 * j) : (16 + 2 * (j - 4))) + 8 * half;
                bf[2 * j]     = br[k];
                bf[2 * j + 1] = br[k + 1];
            }
            acc[t] = __builtin_amdgcn_wmma_f32_16x16x32_bf16(
                false, af, false, bf, (short)0, acc[t], false, false);
        }
        __syncthreads();
    }

#pragma unroll
    for (int t = 0; t < 4; ++t) {
#pragma unroll
        for (int r = 0; r < 8; ++r) {
            int row = m0 + wid * 16 + r + 8 * half;
            int col = n0 + t * 16 + l16;
            float v = acc[t][r];
            if (!(flags & GEMM_ACCUM)) {
                if (bias)  v += bias[col];
                if (bias2) v += bias2[col];
            }
            if (flags & GEMM_RELU) v = fmaxf(v, 0.0f);
            if (flags & GEMM_STORE_BF16) Cb[(size_t)row * N + col] = (__bf16)v;
            else                         Cf[(size_t)row * N + col] = v;
        }
    }
}

// qin = [pos, 0...] so qin @ Wc^T collapses to a rank-2 update.
__global__ void qproj_kernel(const float* __restrict__ pos,
                             const float* __restrict__ Wc,
                             const float* __restrict__ bc,
                             float* __restrict__ qp, int total)
{
    int i = blockIdx.x * blockDim.x + threadIdx.x;
    if (i >= total) return;
    int n = i >> 9, j = i & 511;
    qp[i] = Wc[(size_t)j * 512] * pos[n * 2]
          + Wc[(size_t)j * 512 + 1] * pos[n * 2 + 1] + bc[j];
}

// TDM: async-load a [rows x 128] bf16 tile (row stride 512 elems) into LDS.
// D# per CDNA5 ISA.  pad_interval=64 dwords +
// pad_amount=1 dword => LDS row stride 65 dwords = 130 bf16 (bank-friendly).
__device__ __forceinline__ void tdm_load_k_tile(unsigned lds_addr,
                                                unsigned long long gaddr,
                                                int rows)
{
    tdm_u32x4 g0;
    g0.x = 1u;                                   // count=1, user descriptor
    g0.y = lds_addr;                             // LDS byte address
    g0.z = (unsigned)gaddr;                      // global_addr[31:0]
    g0.w = (unsigned)((gaddr >> 32) & 0x01FFFFFFull) | 0x80000000u; // type=2
    tdm_i32x8 g1;
    g1[0] = (int)((1u << 16)        // data_size = 2 bytes
                | (1u << 20)        // pad_enable
                | (5u << 22));      // pad_interval: 2^(5+1)=64 dwords
                                    // pad_amount=0 -> 1 dword
    g1[1] = (int)(128u << 16);                       // tensor_dim0 = 128
    g1[2] = (int)(((unsigned)rows & 0xFFFFu) << 16); // tensor_dim1 = rows
    g1[3] = (int)(128u << 16);                       // tile_dim0 = 128
    g1[4] = (int)((unsigned)rows & 0xFFFFu);         // tile_dim1 = rows
    g1[5] = 512;                                     // tensor_dim0_stride
    g1[6] = 0;
    g1[7] = 0;
    tdm_i32x4 z4 = {0, 0, 0, 0};
    tdm_i32x8 z8 = {0, 0, 0, 0, 0, 0, 0, 0};
    // 6-arg form (clang-23 / therock-10.0 headers)
    __builtin_amdgcn_tensor_load_to_lds(g0, g1, z4, z4, z8, 0);
}

// One block = one (n, head).  Scores via TDM-staged K tiles (double buffer),
// softmax in LDS, then one coalesced streaming pass over bf16 V.
__global__ __launch_bounds__(128)
void attention_kernel(const float* __restrict__ q,
                      const __bf16* __restrict__ Kc,
                      const __bf16* __restrict__ Vc,
                      float* __restrict__ att, int S)
{
    __shared__ __bf16 Ktile[2][32 * 130];   // 2 x 8.32 KB, TDM-padded rows
    __shared__ float  qs[128];
    __shared__ float  sc[256];
    __shared__ float  psum[128];
    __shared__ float  red[128];

    const int n = blockIdx.y, h = blockIdx.x, tid = threadIdx.x;
    const size_t ebase = ((size_t)n * S) * 512 + (size_t)h * 128;

    qs[tid] = q[(size_t)n * 512 + h * 128 + tid];

    const unsigned long long kgbase =
        (unsigned long long)(const void*)Kc + ebase * 2ull;
    const int ntiles = (S + 31) >> 5;

    // prologue: kick off tile 0 (wave 0 only; scalar branch skips others)
    if (tid < 32) {
        tdm_load_k_tile((unsigned)(size_t)(void*)&Ktile[0][0], kgbase,
                        (S < 32) ? S : 32);
    }

    const float scale = 0.08838834764831845f;  // 1/sqrt(128)
    for (int it = 0; it < ntiles; ++it) {
        if (tid < 32) {
            if (it + 1 < ntiles) {
                int row0 = (it + 1) * 32;
                int rows = S - row0; if (rows > 32) rows = 32;
                tdm_load_k_tile((unsigned)(size_t)(void*)&Ktile[(it + 1) & 1][0],
                                kgbase + (unsigned long long)row0 * 1024ull,
                                rows);
                __builtin_amdgcn_s_wait_tensorcnt(1);  // tile it complete
            } else {
                __builtin_amdgcn_s_wait_tensorcnt(0);  // drain
            }
        }
        __syncthreads();

        // 4 threads per row: thread = (row r, quarter q), 32-wide d-chunks
        const __bf16* tp = &Ktile[it & 1][0];
        const int r = tid >> 2, qq = tid & 3;
        float p = 0.f;
#pragma unroll
        for (int e = 0; e < 32; ++e)
            p += qs[qq * 32 + e] * (float)tp[r * 130 + qq * 32 + e];
        psum[tid] = p;
        __syncthreads();
        if (tid < 32) {
            int row = it * 32 + tid;
            if (row < S)
                sc[row] = (psum[tid * 4] + psum[tid * 4 + 1] +
                           psum[tid * 4 + 2] + psum[tid * 4 + 3]) * scale;
        }
        __syncthreads();   // also protects buffer reuse next iteration
    }

    // ---- softmax over sc[0..S) -----------------------------------------
    float lm = -3.0e38f;
    for (int s = tid; s < S; s += 128) lm = fmaxf(lm, sc[s]);
    red[tid] = lm; __syncthreads();
    for (int o = 64; o > 0; o >>= 1) {
        if (tid < o) red[tid] = fmaxf(red[tid], red[tid + o]);
        __syncthreads();
    }
    float mx = red[0]; __syncthreads();

    float ls = 0.f;
    for (int s = tid; s < S; s += 128) { float e = __expf(sc[s] - mx); sc[s] = e; ls += e; }
    red[tid] = ls; __syncthreads();
    for (int o = 64; o > 0; o >>= 1) {
        if (tid < o) red[tid] += red[tid + o];
        __syncthreads();
    }
    float inv = 1.0f / red[0];
    __syncthreads();
    for (int s = tid; s < S; s += 128) sc[s] *= inv;
    __syncthreads();

    // ---- weighted V: coalesced bf16 stream (256B per step) --------------
    float a = 0.f;
    for (int s = 0; s < S; ++s) {
        const __bf16* vr = Vc + ebase + (size_t)s * 512;
        __builtin_prefetch(vr + 4 * 512, 0, 0);   // global_prefetch_b8
        a += sc[s] * (float)vr[tid];
    }
    att[(size_t)n * 512 + h * 128 + tid] = a;
}

__global__ void role_gather(const int* __restrict__ roles,
                            const float* __restrict__ emb,
                            float* __restrict__ roleN, int total)
{
    int i = blockIdx.x * blockDim.x + threadIdx.x;
    if (i >= total) return;
    int n = i >> 5, k = i & 31;
    roleN[i] = emb[roles[n] * 32 + k];
}

__global__ void build_x(const float* __restrict__ pos,
                        const float* __restrict__ ball,
                        const float* __restrict__ roleN,
                        const float* __restrict__ attO,
                        float* __restrict__ x, int total, int P)
{
    int i = blockIdx.x * blockDim.x + threadIdx.x;
    if (i >= total) return;
    int n = i / 548, j = i - n * 548;
    float v;
    if (j < 2)        v = pos[n * 2 + j];
    else if (j < 4)   v = ball[(n / P) * 2 + (j - 2)];
    else if (j < 36)  v = roleN[n * 32 + (j - 4)];
    else              v = attO[(size_t)n * 512 + (j - 36)];
    x[i] = v;
}

__global__ void lstm_cell(const float* __restrict__ gates,
                          float* __restrict__ h, float* __restrict__ c, int total)
{
    int i = blockIdx.x * blockDim.x + threadIdx.x;
    if (i >= total) return;
    int n = i >> 9, j = i & 511;
    const float* g = gates + (size_t)n * 2048;
    float ig = 1.f / (1.f + __expf(-g[j]));
    float fg = 1.f / (1.f + __expf(-g[512 + j]));
    float gg = tanhf(g[1024 + j]);
    float og = 1.f / (1.f + __expf(-g[1536 + j]));
    float cn = fg * c[i] + ig * gg;
    c[i] = cn;
    h[i] = og * tanhf(cn);
}

__global__ void head_kernel(const float* __restrict__ hid,
                            const float* __restrict__ Wf2,
                            const float* __restrict__ bf2,
                            float* __restrict__ pos, float* __restrict__ out,
                            int Ntot, int T, int t)
{
    int i = blockIdx.x * blockDim.x + threadIdx.x;
    if (i >= Ntot * 2) return;
    int n = i >> 1, o = i & 1;
    const float* hr = hid + (size_t)n * 256;
    const float* wr = Wf2 + o * 256;
    float a = bf2[o];
    for (int k = 0; k < 256; ++k) a += hr[k] * wr[k];
    pos[i] = a;
    out[((size_t)n * T + t) * 2 + o] = a;
}

__global__ void zero_f32(float* p, int total)
{
    int i = blockIdx.x * blockDim.x + threadIdx.x;
    if (i < total) p[i] = 0.f;
}

__global__ void copy_f32(const float* __restrict__ s, float* __restrict__ d, int total)
{
    int i = blockIdx.x * blockDim.x + threadIdx.x;
    if (i < total) d[i] = s[i];
}

static inline dim3 ew_grid(int n) { return dim3((n + 255) / 256); }

extern "C" void kernel_launch(void* const* d_in, const int* in_sizes, int n_in,
                              void* d_out, int out_size, void* d_ws, size_t ws_size,
                              hipStream_t stream)
{
    const float* ctx   = (const float*)d_in[0];
    const float* ipos  = (const float*)d_in[1];
    const float* ball  = (const float*)d_in[2];
    const int*   roles = (const int*)d_in[3];
    // d_in[4] = max_steps device scalar; T derived from out_size (capture-safe)
    const float* emb  = (const float*)d_in[5];
    const float* Wc   = (const float*)d_in[6];  const float* bc   = (const float*)d_in[7];
    const float* Wq   = (const float*)d_in[8];  const float* bq   = (const float*)d_in[9];
    const float* Wk   = (const float*)d_in[10]; const float* bk   = (const float*)d_in[11];
    const float* Wv   = (const float*)d_in[12]; const float* bv   = (const float*)d_in[13];
    const float* Wo   = (const float*)d_in[14]; const float* bo   = (const float*)d_in[15];
    const float* Wih0 = (const float*)d_in[16]; const float* Whh0 = (const float*)d_in[17];
    const float* bih0 = (const float*)d_in[18]; const float* bhh0 = (const float*)d_in[19];
    const float* Wih1 = (const float*)d_in[20]; const float* Whh1 = (const float*)d_in[21];
    const float* bih1 = (const float*)d_in[22]; const float* bhh1 = (const float*)d_in[23];
    const float* Wf1  = (const float*)d_in[24]; const float* bf1  = (const float*)d_in[25];
    const float* Wf2  = (const float*)d_in[26]; const float* bf2  = (const float*)d_in[27];

    const int Ntot = in_sizes[1] / 2;            // 640
    const int S    = in_sizes[0] / (Ntot * 512); // 200
    const int NS   = Ntot * S;                   // 128000
    const int T    = out_size / (Ntot * 2);      // 100
    const int P    = 10;
    const int H    = 512;

    // ---- carve workspace -------------------------------------------------
    char* w = (char*)d_ws;
    auto carve = [&](size_t bytes) -> void* {
        void* p = (void*)w;
        w += (bytes + 255) & ~(size_t)255;
        return p;
    };
    float*  ctxp  = (float*) carve((size_t)NS * H * 4);   // context projection
    __bf16* Kc    = (__bf16*)carve((size_t)NS * H * 2);   // bf16 K cache
    __bf16* Vc    = (__bf16*)carve((size_t)NS * H * 2);   // bf16 V cache
    float*  qp    = (float*) carve((size_t)Ntot * H * 4);
    float*  qbuf  = (float*) carve((size_t)Ntot * H * 4);
    float*  att   = (float*) carve((size_t)Ntot * H * 4);
    float*  attO  = (float*) carve((size_t)Ntot * H * 4);
    float*  xbuf  = (float*) carve((size_t)Ntot * 548 * 4);
    float*  gates = (float*) carve((size_t)Ntot * 2048 * 4);
    float*  hc    = (float*) carve((size_t)4 * Ntot * H * 4);  // h0,c0,h1,c1
    float*  hid   = (float*) carve((size_t)Ntot * 256 * 4);
    float*  pos   = (float*) carve((size_t)Ntot * 2 * 4);
    float*  roleN = (float*) carve((size_t)Ntot * 32 * 4);
    (void)ws_size;

    float* h0 = hc;
    float* c0 = hc + (size_t)Ntot * H;
    float* h1 = hc + (size_t)2 * Ntot * H;
    float* c1 = hc + (size_t)3 * Ntot * H;
    float* out = (float*)d_out;

    const dim3 blk(128);

    // ---- one-time setup --------------------------------------------------
    zero_f32<<<ew_grid(4 * Ntot * H), 256, 0, stream>>>(hc, 4 * Ntot * H);
    copy_f32<<<ew_grid(Ntot * 2), 256, 0, stream>>>(ipos, pos, Ntot * 2);
    role_gather<<<ew_grid(Ntot * 32), 256, 0, stream>>>(roles, emb, roleN, Ntot * 32);

    // ctx_proj = ctx @ Wc^T + bc ; K/V projected and stored bf16
    gemm_wmma<<<dim3(H / TN, NS / TM), blk, 0, stream>>>(
        ctx, Wc, bc, nullptr, ctxp, nullptr, NS, H, 512, 0);
    gemm_wmma<<<dim3(H / TN, NS / TM), blk, 0, stream>>>(
        ctxp, Wk, bk, nullptr, nullptr, Kc, NS, H, H, GEMM_STORE_BF16);
    gemm_wmma<<<dim3(H / TN, NS / TM), blk, 0, stream>>>(
        ctxp, Wv, bv, nullptr, nullptr, Vc, NS, H, H, GEMM_STORE_BF16);

    // ---- T sequential decode steps --------------------------------------
    for (int t = 0; t < T; ++t) {
        qproj_kernel<<<ew_grid(Ntot * H), 256, 0, stream>>>(pos, Wc, bc, qp, Ntot * H);
        gemm_wmma<<<dim3(H / TN, Ntot / TM), blk, 0, stream>>>(
            qp, Wq, bq, nullptr, qbuf, nullptr, Ntot, H, H, 0);

        attention_kernel<<<dim3(4, Ntot), blk, 0, stream>>>(qbuf, Kc, Vc, att, S);

        gemm_wmma<<<dim3(H / TN, Ntot / TM), blk, 0, stream>>>(
            att, Wo, bo, nullptr, attO, nullptr, Ntot, H, H, 0);

        build_x<<<ew_grid(Ntot * 548), 256, 0, stream>>>(pos, ball, roleN, attO,
                                                         xbuf, Ntot * 548, P);
        // layer 0 gates: x @ Wih0^T + b_ih0 + b_hh0, then += h0 @ Whh0^T
        gemm_wmma<<<dim3(2048 / TN, Ntot / TM), blk, 0, stream>>>(
            xbuf, Wih0, bih0, bhh0, gates, nullptr, Ntot, 2048, 548, 0);
        gemm_wmma<<<dim3(2048 / TN, Ntot / TM), blk, 0, stream>>>(
            h0, Whh0, nullptr, nullptr, gates, nullptr, Ntot, 2048, H, GEMM_ACCUM);
        lstm_cell<<<ew_grid(Ntot * H), 256, 0, stream>>>(gates, h0, c0, Ntot * H);

        // layer 1 gates (h0 is new, h1 still old -> correct ordering)
        gemm_wmma<<<dim3(2048 / TN, Ntot / TM), blk, 0, stream>>>(
            h0, Wih1, bih1, bhh1, gates, nullptr, Ntot, 2048, H, 0);
        gemm_wmma<<<dim3(2048 / TN, Ntot / TM), blk, 0, stream>>>(
            h1, Whh1, nullptr, nullptr, gates, nullptr, Ntot, 2048, H, GEMM_ACCUM);
        lstm_cell<<<ew_grid(Ntot * H), 256, 0, stream>>>(gates, h1, c1, Ntot * H);

        gemm_wmma<<<dim3(256 / TN, Ntot / TM), blk, 0, stream>>>(
            h1, Wf1, bf1, nullptr, hid, nullptr, Ntot, 256, H, GEMM_RELU);
        head_kernel<<<ew_grid(Ntot * 2), 256, 0, stream>>>(hid, Wf2, bf2, pos,
                                                           out, Ntot, T, t);
    }
}
